// SubconsciousPlanning_77438260347444
// MI455X (gfx1250) — matrix operand order repaired and checked by reading, not compiled, not run on hardware
//
#include <hip/hip_runtime.h>
#include <hip/hip_bf16.h>
#include <math.h>

typedef __attribute__((ext_vector_type(16))) _Float16 v16h;
typedef __attribute__((ext_vector_type(8)))  float    v8f;

__device__ __forceinline__ float gelu_exact(float x){
  return 0.5f * x * (1.0f + erff(x * 0.70710678118654752f));
}

// A fragment (16x32 f16) from row-major f16 matrix, leading dim lda.
// ISA layout: lanes 0-15 -> M=lane, K {k0..k0+7} in V0-3, {k0+16..k0+23} in V4-7;
//             lanes 16-31 -> M=lane-16, K {k0+8..} and {k0+24..}.
__device__ __forceinline__ v16h load_a_frag(const _Float16* __restrict__ A, int lda,
                                            int row0, int k0, int lane){
  const int r  = row0 + (lane & 15);
  const int kb = k0 + ((lane >> 4) << 3);
  const _Float16* p = A + (size_t)r * lda + kb;
  v16h a;
#pragma unroll
  for (int i = 0; i < 8; ++i) a[i] = p[i];
#pragma unroll
  for (int i = 0; i < 8; ++i) a[8 + i] = p[16 + i];
  return a;
}

// B fragment (32x16 f16) converted on the fly from f32 row-major W (K x N), ld ldw.
// Dense-B layout (matches A): lanes 0-15 hold K=k0..k0+7,k0+16..k0+23 of column n, lanes 16-31 K=k0+8..k0+15,k0+24..k0+31.
__device__ __forceinline__ v16h load_b_frag(const float* __restrict__ W, int ldw,
                                            int k0, int n0, int lane){
  const int n  = n0 + (lane & 15);
  const int kb = k0 + ((lane >> 4) << 3);
  const float* p = W + (size_t)kb * ldw + n;
  v16h b;
#pragma unroll
  for (int i = 0; i < 16; ++i) b[i] = (_Float16)p[(size_t)(i < 8 ? i : i + 8) * ldw];
  return b;
}

// Generic batched GEMM: Out[g, 0..127, n] = act( A[g] @ W[g] + bias[g] ).
// 4 waves / block; wave w owns rows [w*32, w*32+32) -> 2 WMMA accumulators.
// Block = (N-tile of 16) x (batch g); M fixed = 128.
// Weights stream from HBM once (B frags duplicated across the 4 waves hit WGP$/L2).
// ACT: 0 = none, 1 = exact gelu, 2 = tanh * 0.3
template<int ACT, bool OUT_F16>
__global__ __launch_bounds__(128)
void wmma_gemm128(const _Float16* __restrict__ A, int lda, size_t a_bs,
                  const float* __restrict__ W, int ldw, size_t w_bs,
                  const float* __restrict__ bias, size_t b_bs,
                  void* __restrict__ Out, int ldo, size_t o_bs,
                  int K)
{
  const int lane = threadIdx.x & 31;
  const int wv   = threadIdx.x >> 5;           // 0..3
  const int n0 = blockIdx.x * 16;
  const int g  = blockIdx.y;
  A += (size_t)g * a_bs + (size_t)(wv * 32) * lda;
  W += (size_t)g * w_bs;
  const v8f vz = {};
  v8f acc0 = vz, acc1 = vz;

  for (int k0 = 0; k0 < K; k0 += 32){
    v16h bf = load_b_frag(W, ldw, k0, n0, lane);
    v16h a0 = load_a_frag(A, lda, 0,  k0, lane);
    v16h a1 = load_a_frag(A, lda, 16, k0, lane);
    acc0 = __builtin_amdgcn_wmma_f32_16x16x32_f16(false, a0, false, bf,
                                                  (short)0, acc0, false, false);
    acc1 = __builtin_amdgcn_wmma_f32_16x16x32_f16(false, a1, false, bf,
                                                  (short)0, acc1, false, false);
  }

  const int col = n0 + (lane & 15);
  const float bv = bias ? bias[(size_t)g * b_bs + col] : 0.0f;
#pragma unroll
  for (int t = 0; t < 2; ++t){
    const v8f acc = t ? acc1 : acc0;
#pragma unroll
    for (int j = 0; j < 8; ++j){
      const int row = wv * 32 + t * 16 + j + ((lane >> 4) << 3);
      float v = acc[j] + bv;
      if (ACT == 1) v = gelu_exact(v);
      else if (ACT == 2) v = tanhf(v) * 0.3f;
      if (OUT_F16) ((_Float16*)Out)[(size_t)g * o_bs + (size_t)row * ldo + col] = (_Float16)v;
      else         ((float*)Out)    [(size_t)g * o_bs + (size_t)row * ldo + col] = v;
    }
  }
}

// favg[g,b,h] = mean over P=5 of f[g,b,p*2048+h]   (f row stride 10240)
__global__ void favg_kernel(const _Float16* __restrict__ f, _Float16* __restrict__ favg)
{
  const int i = blockIdx.x * 256 + threadIdx.x;     // over 5*128*2048
  if (i >= 5 * 128 * 2048) return;
  const int h  = i & 2047;
  const int gb = i >> 11;
  const _Float16* base = f + (size_t)gb * 10240 + h;
  float s = 0.0f;
#pragma unroll
  for (int p = 0; p < 5; ++p) s += (float)base[p * 2048];
  favg[i] = (_Float16)(s * 0.2f);
}

// scores[v] = concept_field[v,:] . attn_w + attn_b  (per-b offset is constant over v,
// so top-K indices are batch-independent).
__global__ void scores_kernel(const float* __restrict__ cf, const float* __restrict__ aw,
                              const float* __restrict__ ab, float* __restrict__ scores)
{
  const int v = blockIdx.x * 256 + threadIdx.x;
  if (v >= 50257) return;
  const float* row = cf + (size_t)v * 32;
  float s = 0.0f;
#pragma unroll
  for (int n = 0; n < 32; ++n) s += row[n] * aw[n];
  scores[v] = s + ab[0];
}

// Iterative 10-pass global argmax (ties -> lowest index, matching lax.top_k),
// then cmean[n] = mean over the 10 selected concept_field rows.
__global__ void topk_cmean_kernel(const float* __restrict__ scores,
                                  const float* __restrict__ cf,
                                  float* __restrict__ cmean)
{
  __shared__ float smax[256];
  __shared__ int   sidx[256];
  __shared__ int   chosen[10];
  const int t = threadIdx.x;
  for (int k = 0; k < 10; ++k){
    float bestv = -3.0e38f; int besti = 0x7fffffff;
    for (int v = t; v < 50257; v += 256){
      bool skip = false;
      for (int j = 0; j < k; ++j) if (chosen[j] == v) skip = true;
      if (skip) continue;
      const float s = scores[v];
      if (s > bestv || (s == bestv && v < besti)) { bestv = s; besti = v; }
    }
    smax[t] = bestv; sidx[t] = besti;
    __syncthreads();
    for (int off = 128; off > 0; off >>= 1){
      if (t < off){
        if (smax[t + off] > smax[t] ||
            (smax[t + off] == smax[t] && sidx[t + off] < sidx[t])){
          smax[t] = smax[t + off]; sidx[t] = sidx[t + off];
        }
      }
      __syncthreads();
    }
    if (t == 0) chosen[k] = sidx[0];
    __syncthreads();
  }
  if (t < 32){
    float s = 0.0f;
    for (int k = 0; k < 10; ++k) s += cf[(size_t)chosen[k] * 32 + t];
    cmean[t] = s * 0.1f;
  }
}

// ctx[b,n] = cmean[n] + 0.3*(context_embedding[b,:] . ctx_w[:,n] + ctx_b[n])  -> f16
__global__ void ctx_kernel(const float* __restrict__ ce, const float* __restrict__ cw,
                           const float* __restrict__ cb, const float* __restrict__ cmean,
                           _Float16* __restrict__ ctx)
{
  const int b = blockIdx.x, n = threadIdx.x;     // 32 threads
  const float* row = ce + (size_t)b * 2048;
  float s = 0.0f;
  for (int h = 0; h < 2048; ++h) s += row[h] * cw[(size_t)h * 32 + n];
  s += cb[n];
  ctx[b * 32 + n] = (_Float16)(cmean[n] + 0.3f * s);
}

// sc[g,b] = sigmoid(s1[g,b,:] . sc_w2 + sc_b2)
__global__ void sc_kernel(const _Float16* __restrict__ s1, const float* __restrict__ w2,
                          const float* __restrict__ b2, float* __restrict__ sc)
{
  const int g = blockIdx.x, b = threadIdx.x;     // 5 x 128
  const _Float16* row = s1 + ((size_t)g * 128 + b) * 1024;
  float s = 0.0f;
  for (int h = 0; h < 1024; ++h) s += (float)row[h] * w2[h];
  s += b2[0];
  sc[g * 128 + b] = 1.0f / (1.0f + expf(-s));
}

// best = argmax_g sc (first max wins, matching jnp.argmax);
// gather p=0 slice: bf[b,:] = f[best, b, 0:2048]  (f row stride 10240).
__global__ void select_kernel(const float* __restrict__ sc, const _Float16* __restrict__ f,
                              _Float16* __restrict__ bf)
{
  const int b = blockIdx.x;
  float best = sc[b]; int bg = 0;
  for (int g = 1; g < 5; ++g){
    const float v = sc[g * 128 + b];
    if (v > best){ best = v; bg = g; }
  }
  const _Float16* src = f + ((size_t)bg * 128 + b) * 10240;
  for (int h = threadIdx.x; h < 2048; h += blockDim.x) bf[(size_t)b * 2048 + h] = src[h];
}

extern "C" void kernel_launch(void* const* d_in, const int* in_sizes, int n_in,
                              void* d_out, int out_size, void* d_ws, size_t ws_size,
                              hipStream_t stream)
{
  (void)in_sizes; (void)n_in; (void)out_size; (void)ws_size;
  const float* ce   = (const float*)d_in[0];
  const float* cf   = (const float*)d_in[1];
  const float* ctxw = (const float*)d_in[2];
  const float* ctxb = (const float*)d_in[3];
  const float* attw = (const float*)d_in[4];
  const float* attb = (const float*)d_in[5];
  const float* gw1  = (const float*)d_in[6];
  const float* gb1  = (const float*)d_in[7];
  const float* gw2  = (const float*)d_in[8];
  const float* gb2  = (const float*)d_in[9];
  const float* gw3  = (const float*)d_in[10];
  const float* gb3  = (const float*)d_in[11];
  const float* scw1 = (const float*)d_in[12];
  const float* scb1 = (const float*)d_in[13];
  const float* scw2 = (const float*)d_in[14];
  const float* scb2 = (const float*)d_in[15];
  const float* infw = (const float*)d_in[16];
  const float* infb = (const float*)d_in[17];

  char* ws = (char*)d_ws;
  size_t off = 0;
  auto take = [&](size_t bytes)->char*{
    char* p = ws + off; off = (off + bytes + 255) & ~(size_t)255; return p;
  };
  float*    s_scores = (float*)take((size_t)50257 * 4);
  float*    s_cmean  = (float*)take(32 * 4);
  _Float16* s_ctx    = (_Float16*)take((size_t)128 * 32 * 2);
  _Float16* s_h1     = (_Float16*)take((size_t)5 * 128 * 2048 * 2);
  _Float16* s_h2     = (_Float16*)take((size_t)5 * 128 * 2048 * 2);
  _Float16* s_f      = (_Float16*)take((size_t)5 * 128 * 10240 * 2);   // full futures
  _Float16* s_favg   = (_Float16*)take((size_t)5 * 128 * 2048 * 2);
  _Float16* s_s1     = (_Float16*)take((size_t)5 * 128 * 1024 * 2);
  float*    s_sc     = (float*)take(640 * 4);
  _Float16* s_bf     = (_Float16*)take((size_t)128 * 2048 * 2);

  // Stage 1: concept scores + batch-independent top-10 mean + context projection.
  scores_kernel<<<(50257 + 255) / 256, 256, 0, stream>>>(cf, attw, attb, s_scores);
  topk_cmean_kernel<<<1, 256, 0, stream>>>(s_scores, cf, s_cmean);
  ctx_kernel<<<128, 32, 0, stream>>>(ce, ctxw, ctxb, s_cmean, s_ctx);

  // Stage 2: generator MLP via f16 WMMA (f32 accumulate).
  wmma_gemm128<1, true><<<dim3(128, 5), 128, 0, stream>>>(
      s_ctx, 32, 0, gw1, 2048, (size_t)32 * 2048, gb1, 2048,
      s_h1, 2048, (size_t)128 * 2048, 32);
  wmma_gemm128<1, true><<<dim3(128, 5), 128, 0, stream>>>(
      s_h1, 2048, (size_t)128 * 2048, gw2, 2048, (size_t)2048 * 2048, gb2, 2048,
      s_h2, 2048, (size_t)128 * 2048, 2048);
  // Layer 3: P folded into N (N=10240) -> 3200 blocks x 4 waves for HBM saturation.
  wmma_gemm128<0, true><<<dim3(640, 5), 128, 0, stream>>>(
      s_h2, 2048, (size_t)128 * 2048, gw3, 10240, (size_t)2048 * 10240, gb3, 10240,
      s_f, 10240, (size_t)128 * 10240, 2048);
  favg_kernel<<<(5 * 128 * 2048 + 255) / 256, 256, 0, stream>>>(s_f, s_favg);

  // Stage 3: scorer, argmax-select, final influence GEMM.
  wmma_gemm128<1, true><<<dim3(64, 5), 128, 0, stream>>>(
      s_favg, 2048, (size_t)128 * 2048, scw1, 1024, 0, scb1, 0,
      s_s1, 1024, (size_t)128 * 1024, 2048);
  sc_kernel<<<5, 128, 0, stream>>>(s_s1, scw2, scb2, s_sc);
  select_kernel<<<128, 256, 0, stream>>>(s_sc, s_f, s_bf);
  wmma_gemm128<2, false><<<dim3(128, 1), 128, 0, stream>>>(
      s_bf, 2048, 0, infw, 2048, 0, infb, 0,
      (float*)d_out, 2048, 0, 2048);
}
